// ErNET_51075751084337
// MI455X (gfx1250) — 8-bit matrix operand order repaired and checked by reading, compiled, not run on hardware
//
#include <hip/hip_runtime.h>
#include <hip/hip_bf16.h>
#include <stdint.h>

typedef __attribute__((ext_vector_type(8))) int v8i_t;

static __device__ __forceinline__ float fsgn(float v) {
    return (v > 0.f) ? 1.f : ((v < 0.f) ? -1.f : 0.f);
}
static __device__ __forceinline__ int isgn(float v) {
    return (v > 0.f) ? 1 : ((v < 0.f) ? -1 : 0);
}

__global__ void k_conv1(const float* __restrict__ x, const float* __restrict__ w,
                        const float* __restrict__ bias, float* __restrict__ out,
                        int N, int Hin, int Win, int Hout, int Wout) {
    const int NPIX = N * Hout * Wout;
    int idx = blockIdx.x * blockDim.x + threadIdx.x;
    if (idx >= 16 * NPIX) return;
    int oc = idx / NPIX;
    int p  = idx - oc * NPIX;
    int wo = p % Wout;
    int t  = p / Wout;
    int ho = t % Hout;
    int n  = t / Hout;
    float acc = bias[oc];
    for (int ic = 0; ic < 3; ++ic) {
        const float* xb = x + ((size_t)(n * 3 + ic) * Hin) * Win;
        const float* wb = w + (oc * 3 + ic) * 9;
        #pragma unroll
        for (int i = 0; i < 3; ++i) {
            int hi = ho * 2 + i;
            #pragma unroll
            for (int j = 0; j < 3; ++j) {
                int wi = wo * 2 + j;
                acc += xb[hi * Win + wi] * fsgn(wb[i * 3 + j]);
            }
        }
    }
    out[(size_t)oc * NPIX + p] = acc;
}

__global__ void k_dw_pack(const float* __restrict__ act,
                          const float* __restrict__ dw1, const float* __restrict__ db1,
                          const float* __restrict__ dw2, const float* __restrict__ db2,
                          const float* __restrict__ dw3, const float* __restrict__ db3,
                          int8_t* __restrict__ packed,
                          int IC, int N, int Hin, int Win, int Kpad) {
    const int Hout = Hin - 2, Wout = Win - 2;
    const int NPIXout = N * Hout * Wout;
    const int NPIXin  = N * Hin * Win;
    const int c = blockIdx.y;
    int pout = blockIdx.x * blockDim.x + threadIdx.x;
    if (pout >= NPIXout) return;
    int wo = pout % Wout;
    int t  = pout / Wout;
    int ho = t % Hout;
    int n  = t / Hout;
    const float* xb = act + (size_t)c * NPIXin + (size_t)n * Hin * Win;
    const float* dws[3] = {dw1, dw2, dw3};
    const float* dbs[3] = {db1, db2, db3};
    int8_t* prow = packed + (size_t)pout * Kpad;
    #pragma unroll
    for (int br = 0; br < 3; ++br) {
        const int dil = br + 1, pad = br;
        const float* wc = dws[br] + c * 9;
        int acc = 0;
        #pragma unroll
        for (int i = 0; i < 3; ++i) {
            int hi = ho + i * dil - pad;
            if (hi < 0 || hi >= Hin) continue;
            #pragma unroll
            for (int j = 0; j < 3; ++j) {
                int wi = wo + j * dil - pad;
                if (wi < 0 || wi >= Win) continue;
                acc += isgn(wc[i * 3 + j]) * isgn(xb[hi * Win + wi]);
            }
        }
        float v = (float)acc + dbs[br][c];
        prow[br * IC + c] = (int8_t)isgn(v);
    }
    int padK = Kpad - 3 * IC;
    if (c < padK) prow[3 * IC + c] = 0;
}

__global__ void k_pack_w(const float* __restrict__ fw, int8_t* __restrict__ A,
                         int OC, int K, int Kpad) {
    int idx = blockIdx.x * blockDim.x + threadIdx.x;
    if (idx >= OC * Kpad) return;
    int oc = idx / Kpad, k = idx - oc * Kpad;
    A[idx] = (k < K) ? (int8_t)isgn(fw[oc * K + k]) : (int8_t)0;
}

__global__ void k_zero(float* __restrict__ p, int n) {
    int i = blockIdx.x * blockDim.x + threadIdx.x;
    if (i < n) p[i] = 0.f;
}

template <int MT>
__global__ void __launch_bounds__(256) k_gemm_iu8(
    const int8_t* __restrict__ A, const int8_t* __restrict__ B,
    const float* __restrict__ bias, float* __restrict__ out,
    float* __restrict__ stats, int OC, int ocOffset, int Kpad, int NPIX) {
    __shared__ int ldsA[MT * 6 * 32 * 8];

    const int tid  = threadIdx.x;
    const int lane = tid & 31;
    const int wave = tid >> 5;
    const int ocBase0 = ocOffset + (int)blockIdx.y * (MT * 16);
    const int nkb = Kpad >> 6;

    const int* Ag = (const int*)(A + (size_t)ocBase0 * Kpad);
    const int totInts = MT * nkb * 256;
    for (int i = tid; i < totInts; i += 256) {
        int g      = i & 7;
        int lane_t = (i >> 3) & 31;
        int t2     = i >> 8;
        int kblk   = t2 % nkb;
        int mt     = t2 / nkb;
        int m      = lane_t & 15;
        int half_t = lane_t >> 4;
        int kb = (kblk << 6) + ((g >> 2) << 5) + ((g & 3) << 2)
               + (half_t << 4);
        ldsA[i] = Ag[((mt * 16 + m) * Kpad + kb) >> 2];
    }
    __syncthreads();

    const int pixBase = blockIdx.x * 128 + wave * 16;
    const int m    = lane & 15;
    const int half = lane >> 4;
    int pix  = pixBase + m;
    int pixc = (pix < NPIX) ? pix : (NPIX - 1);
    const int8_t* Brow = B + (size_t)pixc * Kpad + (half << 4);

    v8i_t acc[MT];
    #pragma unroll
    for (int mt = 0; mt < MT; ++mt) acc[mt] = (v8i_t){0, 0, 0, 0, 0, 0, 0, 0};

    for (int kblk = 0; kblk < nkb; ++kblk) {
        const int k0 = kblk << 6;
        if (kblk + 1 < nkb)
            __builtin_prefetch(Brow + k0 + 64, 0, 1);
        int4 blo = *(const int4*)(Brow + k0);
        int4 bhi = *(const int4*)(Brow + k0 + 32);
        v8i_t bf = {blo.x, blo.y, blo.z, blo.w, bhi.x, bhi.y, bhi.z, bhi.w};
        #pragma unroll
        for (int mt = 0; mt < MT; ++mt) {
            const v8i_t* ap =
                (const v8i_t*)&ldsA[(((mt * nkb + kblk) << 5) + lane) << 3];
            v8i_t a = *ap;
            acc[mt] = __builtin_amdgcn_wmma_i32_16x16x64_iu8(
                true, a, true, bf, acc[mt], false, false);
        }
    }

    #pragma unroll
    for (int mt = 0; mt < MT; ++mt) {
        #pragma unroll
        for (int g = 0; g < 8; ++g) {
            int oc = ocBase0 + mt * 16 + g + (half << 3);
            float val = (float)acc[mt][g] + bias[oc];
            val = (val >= 0.f) ? val : 0.01f * val;
            int p = pixBase + m;
            if (p < NPIX) out[(size_t)oc * NPIX + p] = val;
            float v1 = (p < NPIX) ? val : 0.f;
            float v2 = v1 * v1;
            #pragma unroll
            for (int o = 8; o >= 1; o >>= 1) {
                v1 += __shfl_xor(v1, o, 32);
                v2 += __shfl_xor(v2, o, 32);
            }
            if (m == 0) {
                atomicAdd(&stats[oc], v1);
                atomicAdd(&stats[OC + oc], v2);
            }
        }
    }
}

__global__ void k_bn(float* __restrict__ buf, const float* __restrict__ stats,
                     const float* __restrict__ gamma, const float* __restrict__ beta,
                     int OC, int NPIX) {
    long long idx = (long long)blockIdx.x * blockDim.x + threadIdx.x;
    if (idx >= (long long)OC * NPIX) return;
    int cch = (int)(idx / NPIX);
    float inv  = 1.f / (float)NPIX;
    float mean = stats[cch] * inv;
    float var  = stats[OC + cch] * inv - mean * mean;
    float r    = rsqrtf(var + 1e-5f);
    float xv   = buf[idx];
    buf[idx] = (xv - mean) * r * gamma[cch] + beta[cch];
}

__global__ void k_maxpool(const float* __restrict__ in, float* __restrict__ out,
                          int C, int N, int H, int W) {
    int Ho = H >> 1, Wo = W >> 1;
    long long total = (long long)C * N * Ho * Wo;
    long long idx = (long long)blockIdx.x * blockDim.x + threadIdx.x;
    if (idx >= total) return;
    int wo = (int)(idx % Wo);
    long long t = idx / Wo;
    int ho = (int)(t % Ho);
    t /= Ho;
    int n = (int)(t % N);
    int cc = (int)(t / N);
    const float* ib = in + (((size_t)(cc * N + n) * H + 2 * ho) * W + 2 * wo);
    float m0 = fmaxf(ib[0], ib[1]);
    float m1 = fmaxf(ib[W], ib[W + 1]);
    out[idx] = fmaxf(m0, m1);
}

__global__ void __launch_bounds__(256) k_final(
    const float* __restrict__ act,
    const float* __restrict__ w2, const float* __restrict__ b2,
    const float* __restrict__ fcw, const float* __restrict__ fcb,
    float* __restrict__ out, int N) {
    __shared__ float conv2s[5 * 49];
    __shared__ float pooled[45];
    __shared__ float logits[5];
    const int n = blockIdx.x;
    const int tid = threadIdx.x;
    const int NP = N * 49;
    if (tid < 245) {
        int oc = tid / 49, pix = tid % 49;
        int acc = 0;
        for (int ic = 0; ic < 256; ++ic)
            acc += isgn(act[(size_t)ic * NP + n * 49 + pix]) * isgn(w2[oc * 256 + ic]);
        conv2s[tid] = (float)acc + b2[oc];
    }
    __syncthreads();
    if (tid < 45) {
        int oc = tid / 9, cell = tid % 9;
        int oh = cell / 3, ow = cell % 3;
        float s = 0.f;
        #pragma unroll
        for (int i = 0; i < 5; ++i)
            #pragma unroll
            for (int j = 0; j < 5; ++j)
                s += conv2s[oc * 49 + (oh + i) * 7 + (ow + j)];
        pooled[tid] = s * (1.f / 25.f);
    }
    __syncthreads();
    if (tid < 5) {
        int acc = 0;
        for (int j = 0; j < 45; ++j)
            acc += isgn(pooled[j]) * isgn(fcw[tid * 45 + j]);
        logits[tid] = (float)acc + fcb[tid];
    }
    __syncthreads();
    if (tid == 0) {
        float mx = logits[0];
        for (int k = 1; k < 5; ++k) mx = fmaxf(mx, logits[k]);
        float s = 0.f, e[5];
        for (int k = 0; k < 5; ++k) { e[k] = __expf(logits[k] - mx); s += e[k]; }
        for (int k = 0; k < 5; ++k) out[n * 5 + k] = e[k] / s;
    }
}

extern "C" void kernel_launch(void* const* d_in, const int* in_sizes, int n_in,
                              void* d_out, int out_size, void* d_ws, size_t ws_size,
                              hipStream_t stream) {
    (void)in_sizes; (void)n_in; (void)out_size; (void)ws_size;
    const float* x   = (const float*)d_in[0];
    const float* c1w = (const float*)d_in[1];
    const float* c1b = (const float*)d_in[2];
    const float *dw1[6], *db1[6], *dw2[6], *db2[6], *dw3[6], *db3[6],
                *fw[6], *fb[6], *gamma[6], *beta[6];
    for (int s = 0; s < 6; ++s) {
        int b = 3 + s * 10;
        dw1[s] = (const float*)d_in[b + 0]; db1[s] = (const float*)d_in[b + 1];
        dw2[s] = (const float*)d_in[b + 2]; db2[s] = (const float*)d_in[b + 3];
        dw3[s] = (const float*)d_in[b + 4]; db3[s] = (const float*)d_in[b + 5];
        fw[s]  = (const float*)d_in[b + 6]; fb[s]  = (const float*)d_in[b + 7];
        gamma[s] = (const float*)d_in[b + 8]; beta[s] = (const float*)d_in[b + 9];
    }
    const float* w2  = (const float*)d_in[63];
    const float* b2  = (const float*)d_in[64];
    const float* fcw = (const float*)d_in[65];
    const float* fcb = (const float*)d_in[66];
    float* outp = (float*)d_out;

    const int N = 64;
    const int ICs[6]   = {16, 64, 96, 128, 128, 128};
    const int OCs[6]   = {64, 96, 128, 128, 128, 256};
    const int Kpads[6] = {64, 192, 320, 384, 384, 384};
    const int Hin[6]   = {118, 58, 28, 13, 11, 9};

    uint8_t* ws = (uint8_t*)d_ws;
    size_t off = 0;
    auto carve = [&](size_t bytes) {
        uint8_t* p = ws + off;
        off += (bytes + 255) & ~(size_t)255;
        return p;
    };
    float*  r0 = (float*)carve((size_t)16 * N * 118 * 118 * 4);
    float*  r1 = (float*)carve((size_t)64 * N * 116 * 116 * 4);
    int8_t* P  = (int8_t*)carve((size_t)N * 116 * 116 * 64);
    float*  stats = (float*)carve(2 * 256 * sizeof(float));
    int8_t* Apk[6];
    for (int s = 0; s < 6; ++s) Apk[s] = (int8_t*)carve((size_t)OCs[s] * Kpads[s]);

    {
        int tot = 16 * N * 118 * 118;
        k_conv1<<<(tot + 255) / 256, 256, 0, stream>>>(x, c1w, c1b, r0,
                                                       N, 237, 237, 118, 118);
    }
    for (int s = 0; s < 6; ++s) {
        int tot = OCs[s] * Kpads[s];
        k_pack_w<<<(tot + 255) / 256, 256, 0, stream>>>(fw[s], Apk[s], OCs[s],
                                                        3 * ICs[s], Kpads[s]);
    }

    float* bufs_in[6]  = {r0, r0, r0, r0, r1, r0};
    float* bufs_out[6] = {r1, r1, r1, r1, r0, r1};
    for (int s = 0; s < 6; ++s) {
        const int IC = ICs[s], OC = OCs[s], Kpad = Kpads[s];
        const int Hi = Hin[s], Wi = Hi, Ho = Hi - 2, Wo = Ho;
        const int NPIXout = N * Ho * Wo;
        dim3 g1((NPIXout + 255) / 256, IC);
        k_dw_pack<<<g1, 256, 0, stream>>>(bufs_in[s], dw1[s], db1[s], dw2[s],
                                          db2[s], dw3[s], db3[s], P,
                                          IC, N, Hi, Wi, Kpad);
        k_zero<<<(2 * OC + 255) / 256, 256, 0, stream>>>(stats, 2 * OC);
        const int xBlocks = (NPIXout + 127) / 128;
        const int yFull   = OC / 64;
        if (yFull > 0) {
            dim3 g2(xBlocks, yFull);
            k_gemm_iu8<4><<<g2, 256, 0, stream>>>(Apk[s], P, fb[s], bufs_out[s],
                                                  stats, OC, 0, Kpad, NPIXout);
        }
        if (OC % 64) {
            dim3 g2r(xBlocks, 1);
            k_gemm_iu8<2><<<g2r, 256, 0, stream>>>(Apk[s], P, fb[s], bufs_out[s],
                                                   stats, OC, yFull * 64, Kpad,
                                                   NPIXout);
        }
        long long tot = (long long)OC * NPIXout;
        k_bn<<<(unsigned)((tot + 255) / 256), 256, 0, stream>>>(bufs_out[s], stats,
                                                                gamma[s], beta[s],
                                                                OC, NPIXout);
        if (s < 3) {
            long long tp = (long long)OC * N * (Ho >> 1) * (Wo >> 1);
            k_maxpool<<<(unsigned)((tp + 255) / 256), 256, 0, stream>>>(
                bufs_out[s], r0, OC, N, Ho, Wo);
        }
    }
    k_final<<<N, 256, 0, stream>>>(r1, w2, b2, fcw, fcb, outp, N);
}
